// VectorQuantizer_35278861369466
// MI455X (gfx1250) — matrix operand order repaired and checked by reading, not compiled, not run on hardware
//
#include <hip/hip_runtime.h>

// ---------------------------------------------------------------------------
// VectorQuantizer for MI455X (gfx1250):
//   argmin_k ( c_sq[k] - 2 * <x[n], codebook[k]> )        (x_sq dropped: const/row)
// cross term via split-bf16 WMMA: x = xh + xl, c = ch + cl,
//   x.c ~= xh.ch + xh.cl + xl.ch   (f32 accumulate, ~2^-16 rel err)
//
// Launch shape: 256 WGs x 256 thr (8 waves). Waves 0-3 own 128 rows and
// K-tiles [0,256); waves 4-7 own the same rows and K-tiles [256,512).
// Final argmin combine across the two K-halves goes through 2 KB of LDS.
// ---------------------------------------------------------------------------

typedef __bf16 v16bf __attribute__((ext_vector_type(16)));
typedef float  v8f   __attribute__((ext_vector_type(8)));

#define VQ_N 32768
#define VQ_K 8192
#define VQ_D 64
#define VQ_KTILES (VQ_K / 16)       // 512
#define VQ_KTHALF (VQ_KTILES / 2)   // 256 tiles per K-half

// float -> bf16 round-to-nearest-even, via bit ops (no reliance on __bf16 arith)
__device__ __forceinline__ __bf16 f2bf(float f) {
  unsigned u = __builtin_bit_cast(unsigned, f);
  unsigned r = (u + 0x7FFFu + ((u >> 16) & 1u)) >> 16;
  unsigned short s = (unsigned short)r;
  return __builtin_bit_cast(__bf16, s);
}
__device__ __forceinline__ float bf2f(__bf16 b) {
  unsigned short s = __builtin_bit_cast(unsigned short, b);
  unsigned u = ((unsigned)s) << 16;
  return __builtin_bit_cast(float, u);
}

// ---------------------------------------------------------------------------
// Pre-pass 1: c_sq[k] = sum_d codebook[k][d]^2  (exact f32)
// ---------------------------------------------------------------------------
__global__ void vq_csq_kernel(const float* __restrict__ cb, float* __restrict__ csq) {
  int k = blockIdx.x * blockDim.x + threadIdx.x;
  if (k >= VQ_K) return;
  const float* r = cb + (size_t)k * VQ_D;
  float s = 0.f;
#pragma unroll
  for (int j = 0; j < VQ_D; ++j) s += r[j] * r[j];
  csq[k] = s;
}

// ---------------------------------------------------------------------------
// Pre-pass 2: pack codebook into WMMA B-matrix per-lane layout, bf16 hi + lo.
// Layout: bpack[(((kt*2 + chunk)*2 + mat)*32 + lane)*16 + j]
//   lane = column n (mod 16), lanes 16..31 carry K=16..31 of the 32-deep chunk
//   B element (K, n): K = (lane<16 ? 0 : 16) + j + 32*chunk   (this code's order, same as A)
// ---------------------------------------------------------------------------
__global__ void vq_pack_kernel(const float* __restrict__ cb, __bf16* __restrict__ bpack) {
  int tid = blockIdx.x * blockDim.x + threadIdx.x;
  if (tid >= VQ_KTILES * 32) return;
  int kt   = tid >> 5;
  int lane = tid & 31;
  int n    = lane & 15;
  int koff = (lane < 16) ? 0 : 16;
  const float* row = cb + (size_t)(kt * 16 + n) * VQ_D;
#pragma unroll
  for (int c = 0; c < 2; ++c) {
    size_t base_hi = ((size_t)((kt * 2 + c) * 2 + 0) * 32 + lane) * 16;
    size_t base_lo = ((size_t)((kt * 2 + c) * 2 + 1) * 32 + lane) * 16;
#pragma unroll
    for (int j = 0; j < 16; ++j) {
      float v  = row[c * 32 + koff + j];
      __bf16 h = f2bf(v);
      __bf16 l = f2bf(v - bf2f(h));
      bpack[base_hi + j] = h;
      bpack[base_lo + j] = l;
    }
  }
}

// ---------------------------------------------------------------------------
// Main kernel.
// ---------------------------------------------------------------------------
__global__ __launch_bounds__(256) void vq_main_kernel(
    const float*  __restrict__ x,
    const __bf16* __restrict__ bpack,
    const float*  __restrict__ csq,
    int*          __restrict__ out)
{
  __shared__ float s_val[2][128];
  __shared__ int   s_idx[2][128];

  const int lane  = threadIdx.x & 31;
  // Wave index is uniform across the wave: pin it to an SGPR so the K-loop
  // gets scalar (s_cmp) control flow and SGPR-base addressing, not an
  // exec-masked divergent loop with 64-bit VALU address math.
  const int wave  = __builtin_amdgcn_readfirstlane(threadIdx.x >> 5);
  const int rgrp  = wave & 3;        // row group within WG (0..3) -> 32 rows each
  const int khalf = wave >> 2;       // 0: K-tiles [0,256), 1: [256,512)
  const int hhalf = lane >> 4;       // 0: lanes 0-15, 1: lanes 16-31
  const int lm    = lane & 15;
  const int wgRow   = blockIdx.x * 128;        // 128 rows per WG
  const int rowBase = wgRow + rgrp * 32;       // 32 rows per wave

  // ---- Load A (x rows) once, split into bf16 hi/lo in WMMA A layout ----
  // A 16x32 bf16 layout: lane m holds, for j=0..15:
  //   K = j + 16*laneHalf + 32*chunk
  v16bf Ahi[2][2], Alo[2][2];   // [tile][chunk]
#pragma unroll
  for (int t = 0; t < 2; ++t) {
#pragma unroll
    for (int c = 0; c < 2; ++c) {
      const float* xr = x + (size_t)(rowBase + t * 16 + lm) * VQ_D + hhalf * 16 + c * 32;
#pragma unroll
      for (int j = 0; j < 8; ++j) {
        float v0 = xr[j];        // K = c*32 + hhalf*16 + j
        float v1 = xr[8 + j];    // K = c*32 + hhalf*16 + 8 + j
        __bf16 h0 = f2bf(v0), h1 = f2bf(v1);
        Ahi[t][c][j]     = h0;  Alo[t][c][j]     = f2bf(v0 - bf2f(h0));
        Ahi[t][c][8 + j] = h1;  Alo[t][c][8 + j] = f2bf(v1 - bf2f(h1));
      }
    }
  }

  float minv[2][8];
  int   mini[2][8];
#pragma unroll
  for (int t = 0; t < 2; ++t)
#pragma unroll
    for (int v = 0; v < 8; ++v) { minv[t][v] = 3.4e38f; mini[t][v] = 0; }

  // ---- K loop: 256 tiles of 16 codes (this wave's half of K) ----
  const int ktBeg = khalf * VQ_KTHALF;
  const int ktEnd = ktBeg + VQ_KTHALF;
  for (int kt = ktBeg; kt < ktEnd; ++kt) {
    // Prefetch packed B two tiles ahead (4 KB/tile; one cacheline per lane).
    if (kt + 2 < ktEnd) {
      const char* pf = (const char*)(bpack + (size_t)(kt + 2) * 4 * 32 * 16);
      __builtin_prefetch(pf + lane * 128, 0, 0);
    }

    // B tiles: straight 32B per lane per matrix (pre-packed layout).
    size_t tb = (size_t)kt * 4 * 32 * 16;
    v16bf Bhi0 = *(const v16bf*)(bpack + tb + (size_t)(0 * 32 + lane) * 16);
    v16bf Blo0 = *(const v16bf*)(bpack + tb + (size_t)(1 * 32 + lane) * 16);
    v16bf Bhi1 = *(const v16bf*)(bpack + tb + (size_t)(2 * 32 + lane) * 16);
    v16bf Blo1 = *(const v16bf*)(bpack + tb + (size_t)(3 * 32 + lane) * 16);

    float cs  = csq[kt * 16 + lm];
    int   idx = kt * 16 + lm;

#pragma unroll
    for (int t = 0; t < 2; ++t) {
      v8f a = {};
      // hi*hi (both D-chunks), then cross corrections; lo*lo dropped.
      a = __builtin_amdgcn_wmma_f32_16x16x32_bf16(false, Ahi[t][0], false, Bhi0, (short)0, a, false, false);
      a = __builtin_amdgcn_wmma_f32_16x16x32_bf16(false, Ahi[t][1], false, Bhi1, (short)0, a, false, false);
      a = __builtin_amdgcn_wmma_f32_16x16x32_bf16(false, Ahi[t][0], false, Blo0, (short)0, a, false, false);
      a = __builtin_amdgcn_wmma_f32_16x16x32_bf16(false, Ahi[t][1], false, Blo1, (short)0, a, false, false);
      a = __builtin_amdgcn_wmma_f32_16x16x32_bf16(false, Alo[t][0], false, Bhi0, (short)0, a, false, false);
      a = __builtin_amdgcn_wmma_f32_16x16x32_bf16(false, Alo[t][1], false, Bhi1, (short)0, a, false, false);

      // Running argmin: lane holds column n=lm, rows M = v + 8*hhalf.
#pragma unroll
      for (int v = 0; v < 8; ++v) {
        float d = __builtin_fmaf(-2.f, a[v], cs);   // c_sq - 2*cross
        if (d < minv[t][v]) { minv[t][v] = d; mini[t][v] = idx; }
      }
    }
  }

  // ---- Cross-lane min-reduce over the 16 columns (stays within lane half),
  //      tie-break to smallest index (jnp.argmin semantics). ----
#pragma unroll
  for (int t = 0; t < 2; ++t) {
#pragma unroll
    for (int v = 0; v < 8; ++v) {
      float bv = minv[t][v];
      int   bi = mini[t][v];
#pragma unroll
      for (int off = 1; off < 16; off <<= 1) {
        float ov = __shfl_xor(bv, off, 32);
        int   oi = __shfl_xor(bi, off, 32);
        if (ov < bv || (ov == bv && oi < bi)) { bv = ov; bi = oi; }
      }
      if (lm == 0) {
        int r = rgrp * 32 + t * 16 + v + hhalf * 8;   // row within WG [0,128)
        s_val[khalf][r] = bv;
        s_idx[khalf][r] = bi;
      }
    }
  }

  __syncthreads();

  // ---- Combine the two K-halves; half 0 has smaller k, so ties pick it. ----
  if (threadIdx.x < 128) {
    int r = threadIdx.x;
    float v0 = s_val[0][r], v1 = s_val[1][r];
    int   i0 = s_idx[0][r], i1 = s_idx[1][r];
    out[wgRow + r] = (v0 <= v1) ? i0 : i1;
  }
}

// ---------------------------------------------------------------------------
extern "C" void kernel_launch(void* const* d_in, const int* in_sizes, int n_in,
                              void* d_out, int out_size, void* d_ws, size_t ws_size,
                              hipStream_t stream) {
  const float* x  = (const float*)d_in[0];   // [N, D] f32
  const float* cb = (const float*)d_in[1];   // [K, D] f32
  int* out = (int*)d_out;                    // [N] int32 argmin indices

  float*  csq   = (float*)d_ws;                                  // 32 KB
  __bf16* bpack = (__bf16*)((char*)d_ws + VQ_K * sizeof(float)); // 2 MB packed B

  vq_csq_kernel <<<VQ_K / 256, 256, 0, stream>>>(cb, csq);
  vq_pack_kernel<<<(VQ_KTILES * 32) / 256, 256, 0, stream>>>(cb, bpack);
  vq_main_kernel<<<VQ_N / 128, 256, 0, stream>>>(x, bpack, csq, out);
}
